// G2Braph_20469814133058
// MI455X (gfx1250) — matrix operand order repaired and checked by reading, not compiled, not run on hardware
//
#include <hip/hip_runtime.h>
#include <hip/hip_bf16.h>
#include <math.h>

#define N_NODES 20000
#define N_EDGES 320000
#define D       50
#define TOWERS  5
#define F_OUT   10
#define N_LAYERS 4

typedef __attribute__((ext_vector_type(16))) _Float16 v16h;
typedef __attribute__((ext_vector_type(8)))  _Float16 v8h;
typedef __attribute__((ext_vector_type(8)))  float    v8f;
typedef __attribute__((ext_vector_type(4))) unsigned int u32x4;
typedef __attribute__((ext_vector_type(8))) int i32x8;
typedef __attribute__((ext_vector_type(4))) int i32x4;

#if __has_builtin(__builtin_amdgcn_tensor_load_to_lds) && __has_builtin(__builtin_amdgcn_s_wait_tensorcnt)
#define HAVE_TDM 1
#else
#define HAVE_TDM 0
#endif

// ---------- WMMA fragment helpers ----------
// A fragment (16x32 f16, row-major, stride in halves).
// Lane L: M = L%16; halves[0..7] = A[M][k0+8*hi .. +7], halves[8..15] = A[M][k0+16+8*hi .. +7]
static __device__ __forceinline__ v16h load_a_frag(const _Float16* base, int stride_h, int k0, int lane) {
  const int m = lane & 15, hi = lane >> 4;
  const _Float16* p = base + m * stride_h + k0 + 8 * hi;
  v8h lo = *reinterpret_cast<const v8h*>(p);
  v8h hb = *reinterpret_cast<const v8h*>(p + 16);
  v16h a;
#pragma unroll
  for (int i = 0; i < 8; ++i) { a[i] = lo[i]; a[i + 8] = hb[i]; }
  return a;
}
// B fragment (32x16 f16) from pre-transposed Bt[n][k] (N-major, contiguous K).
static __device__ __forceinline__ v16h load_b_frag(const _Float16* bt, int stride_h, int k0, int lane) {
  const int n = lane & 15, hi = lane >> 4;
  const _Float16* p = bt + n * stride_h + k0 + 8 * hi;
  v8h lo = *reinterpret_cast<const v8h*>(p);
  v8h hb = *reinterpret_cast<const v8h*>(p + 16);
  v16h b;
#pragma unroll
  for (int i = 0; i < 8; ++i) { b[i] = lo[i]; b[i + 8] = hb[i]; }
  return b;
}
#define WMMA_F16(a, b, c) \
  __builtin_amdgcn_wmma_f32_16x16x32_f16(false, (a), false, (b), (short)0, (c), false, false)

// ---------- init / prep ----------
__global__ void k_zero_f32(float* p, int n) {
  int i = blockIdx.x * 256 + threadIdx.x;
  if (i < n) p[i] = 0.f;
}
__global__ void k_degree(const int* ei, float* deg) {
  int e = blockIdx.x * 256 + threadIdx.x;
  if (e < N_EDGES) atomicAdd(&deg[ei[N_EDGES + e]], 1.f);
}
__global__ void k_avglog(const float* deg, float* s) {
  __shared__ float red[256];
  int n = blockIdx.x * 256 + threadIdx.x;
  red[threadIdx.x] = (n < N_NODES) ? logf(deg[n] + 1.f) : 0.f;
  __syncthreads();
  for (int st = 128; st > 0; st >>= 1) {
    if (threadIdx.x < st) red[threadIdx.x] += red[threadIdx.x + st];
    __syncthreads();
  }
  if (threadIdx.x == 0) atomicAdd(s, red[0]);
}
// exclusive scan of deg -> rowptr (and cursor copy); single block of 256 threads
__global__ void k_scan_rowptr(const float* deg, int* rowptr, int* cursor) {
  const int PER = (N_NODES + 255) / 256;  // 79
  __shared__ int part[256];
  __shared__ int partx[257];
  int t = threadIdx.x;
  int s = 0;
  for (int i = 0; i < PER; ++i) {
    int n = t * PER + i;
    if (n < N_NODES) s += (int)deg[n];
  }
  part[t] = s;
  __syncthreads();
  if (t == 0) {
    int acc = 0;
    for (int i = 0; i < 256; ++i) { partx[i] = acc; acc += part[i]; }
    partx[256] = acc;
  }
  __syncthreads();
  int acc = partx[t];
  for (int i = 0; i < PER; ++i) {
    int n = t * PER + i;
    if (n < N_NODES) { rowptr[n] = acc; cursor[n] = acc; acc += (int)deg[n]; }
  }
  if (t == 0) rowptr[N_NODES] = partx[256];
}
__global__ void k_build_csr(const int* ei, int* cursor, int* csr_src) {
  int e = blockIdx.x * 256 + threadIdx.x;
  if (e < N_EDGES) {
    int pos = atomicAdd(&cursor[ei[N_EDGES + e]], 1);
    csr_src[pos] = ei[e];
  }
}
__global__ void k_embed(const int* x, const float* emb, float* h, _Float16* hf16) {
  int i = blockIdx.x * 256 + threadIdx.x;
  if (i >= N_NODES * 64) return;
  int n = i >> 6, c = i & 63;
  float v = (c < D) ? emb[x[n] * D + c] : 0.f;
  h[i] = v; hf16[i] = (_Float16)v;
}
// Bpre: [L][512][64] f16 (cols 0..249 = x_i part, 256..505 = x_j part; k = f)
__global__ void k_prep_pre(const float* pre_W, _Float16* Bpre) {
  int i = blockIdx.x * 256 + threadIdx.x;
  if (i >= N_LAYERS * 512 * 64) return;
  int f = i & 63, c = (i >> 6) & 511, l = i >> 15;
  float v = 0.f;
  if (f < D) {
    if (c < TOWERS * D) {
      int t = c / D, g = c % D;
      v = pre_W[((l * TOWERS + t) * D + g) * (2 * D) + f];
    } else if (c >= 256 && c < 256 + TOWERS * D) {
      int cc = c - 256, t = cc / D, g = cc % D;
      v = pre_W[((l * TOWERS + t) * D + g) * (2 * D) + D + f];
    }
  }
  Bpre[i] = (_Float16)v;
}
// Bpost: [L][T][16][672] f16 (n-major, contiguous K; rows >= F_OUT and k >= 650 zero)
__global__ void k_prep_post(const float* post_W, _Float16* Bpost) {
  int i = blockIdx.x * 256 + threadIdx.x;
  if (i >= N_LAYERS * TOWERS * 16 * 672) return;
  int k = i % 672, r = i / 672;
  int g = r % 16; r /= 16;
  int t = r % TOWERS, l = r / TOWERS;
  float v = 0.f;
  if (g < F_OUT && k < 13 * D) v = post_W[((l * TOWERS + t) * F_OUT + g) * (13 * D) + k];
  Bpost[i] = (_Float16)v;
}
// Blin: [L][64][64] f16, Bt[o][i] = lin_W[l][o][i]
__global__ void k_prep_lin(const float* lin_W, _Float16* Blin) {
  int i = blockIdx.x * 256 + threadIdx.x;
  if (i >= N_LAYERS * 64 * 64) return;
  int ii = i & 63, o = (i >> 6) & 63, l = i >> 12;
  float v = (o < D && ii < D) ? lin_W[(l * D + o) * D + ii] : 0.f;
  Blin[i] = (_Float16)v;
}

// ---------- GEMM: A[N,64] f16 x Bt[ncols,64] f16 -> out f32; 4 N-tiles per wave ----------
__global__ void k_gemm_nk64(const _Float16* __restrict__ A, const _Float16* __restrict__ Bt,
                            float* __restrict__ out, int out_stride,
                            const float* __restrict__ bias, int bias_n) {
  int lane = threadIdx.x;
  int mt = blockIdx.x;
  int nt0 = blockIdx.y * 4;
  v16h a0 = load_a_frag(A + mt * 16 * 64, 64, 0, lane);
  v16h a1 = load_a_frag(A + mt * 16 * 64, 64, 32, lane);
  int rbase = mt * 16 + 8 * (lane >> 4);
#pragma unroll
  for (int j = 0; j < 4; ++j) {
    int nt = nt0 + j;
    v8f acc = {};
    acc = WMMA_F16(a0, load_b_frag(Bt + nt * 16 * 64, 64, 0, lane), acc);
    acc = WMMA_F16(a1, load_b_frag(Bt + nt * 16 * 64, 64, 32, lane), acc);
    int col = nt * 16 + (lane & 15);
    float bv = (bias != nullptr && col < bias_n) ? bias[col] : 0.f;
#pragma unroll
    for (int v = 0; v < 8; ++v) out[(rbase + v) * out_stride + col] = acc[v] + bv;
  }
}

// ---------- CSR aggregation: one block per node; register min/max/sum/sumsq, no atomics ----------
__global__ void k_aggregate(const int* __restrict__ rowptr, const int* __restrict__ csr_src,
                            const float* __restrict__ aiaj, const float* __restrict__ preb,
                            float* __restrict__ ssum, float* __restrict__ ssq,
                            float* __restrict__ smn, float* __restrict__ smx) {
  __shared__ int srcs[256];
  int n = blockIdx.x;
  int c = threadIdx.x;
  int begin = rowptr[n], end = rowptr[n + 1];
  float aid = (c < TOWERS * D) ? (aiaj[n * 512 + c] + preb[c]) : 0.f;
  float sum = 0.f, sq = 0.f;
  float mn = 3.402823466e+38f, mx = -3.402823466e+38f;
  for (int chunk = begin; chunk < end; chunk += 256) {
    int e = chunk + threadIdx.x;
    if (e < end) srcs[threadIdx.x] = csr_src[e];
    __syncthreads();
    int cnt = min(256, end - chunk);
    for (int i = 0; i < cnt; ++i) {
      float m = aid + aiaj[srcs[i] * 512 + 256 + c];
      sum += m; sq += m * m;
      mn = fminf(mn, m); mx = fmaxf(mx, m);
    }
    __syncthreads();
  }
  if (c < TOWERS * D) {
    int o = n * 256 + c;
    bool nonempty = end > begin;
    ssum[o] = sum; ssq[o] = sq;
    smn[o] = nonempty ? mn : 0.f;
    smx[o] = nonempty ? mx : 0.f;
  }
}

// ---------- finalize: PNA features in LDS per 16 nodes; Bpost tile via TDM; WMMA post-NN ----------
__global__ void k_finalize(const float* __restrict__ h, const float* __restrict__ ssum,
                           const float* __restrict__ ssq, const float* __restrict__ smn,
                           const float* __restrict__ smx, const float* __restrict__ deg,
                           const float* __restrict__ avgl, const _Float16* __restrict__ Bpost,
                           const float* __restrict__ postb, _Float16* __restrict__ yf16) {
  __shared__ __align__(64) _Float16 smem[2 * 16 * 672];
  _Float16* feat = smem;                 // LDS offset 0
  _Float16* bsh = smem + 16 * 672;       // LDS byte offset 21504
  int lane = threadIdx.x;
  int tile = blockIdx.x;
  float avg_log = avgl[0] * (1.f / N_NODES);
  for (int t = 0; t < TOWERS; ++t) {
#if HAVE_TDM
    {
      // 2D TDM descriptor: 16 rows x 672 halves, global -> LDS @21504, overlaps feature build
      unsigned long long ga = (unsigned long long)(const void*)(Bpost + t * 16 * 672);
      u32x4 g0;
      g0[0] = 1u;                                                 // count=1
      g0[1] = (unsigned)(2 * 16 * 672);                           // lds_addr (bytes) = 21504
      g0[2] = (unsigned)(ga & 0xFFFFFFFFu);                       // global_addr[31:0]
      g0[3] = (unsigned)((ga >> 32) & 0x1FFFFFFu) | (2u << 30);   // global_addr[56:32] | type=2
      i32x8 g1;
      g1[0] = 1 << 16;                   // workgroup_mask=0, data_size=1 (2 bytes)
      g1[1] = (672 & 0xFFFF) << 16;      // tensor_dim0[15:0] @ bits 63:48
      g1[2] = (16 << 16);                // tensor_dim0[31:16]=0, tensor_dim1[15:0]=16
      g1[3] = (672 << 16);               // tensor_dim1[31:16]=0, tile_dim0=672
      g1[4] = 16;                        // tile_dim1=16, tile_dim2=0
      g1[5] = 672;                       // tensor_dim0_stride[31:0]
      g1[6] = 0;                         // stride hi, tensor_dim1_stride lo
      g1[7] = 0;
      i32x4 zz = {0, 0, 0, 0};
      i32x8 z8 = {0, 0, 0, 0, 0, 0, 0, 0};
      __builtin_amdgcn_tensor_load_to_lds(g0, g1, zz, zz, z8, 0);
    }
#else
    for (int idx = lane; idx < 16 * 672 / 8; idx += 32)
      reinterpret_cast<v8h*>(bsh)[idx] = reinterpret_cast<const v8h*>(Bpost + t * 16 * 672)[idx];
#endif
    // build 16 x 672 feature block (650 valid) while the TDM runs
    for (int idx = lane; idx < 16 * 672; idx += 32) {
      int nl = idx / 672, k = idx % 672;
      int node = tile * 16 + nl;
      float v = 0.f;
      if (k < D) {
        v = h[node * 64 + k];                          // x_t identity part
      } else if (k < 13 * D) {
        int j = k - D;
        int grp = j / (4 * D), r = j % (4 * D), s = r / D, g = r % D;
        int o = node * 256 + t * D + g;
        float cnt = fmaxf(deg[node], 1.f);
        float stat;
        if (s == 0) {
          stat = ssum[o] / cnt;                        // mean
        } else if (s == 1) {
          stat = smn[o];                               // min (0 for empty)
        } else if (s == 2) {
          stat = smx[o];                               // max (0 for empty)
        } else {
          float mean = ssum[o] / cnt;
          float var = ssq[o] / cnt - mean * mean;
          stat = sqrtf(fmaxf(var, 0.f) + 1e-5f);       // std
        }
        float sc = logf(cnt + 1.f) / avg_log;
        if (grp == 1) stat *= sc;                      // amplification
        else if (grp == 2) stat /= sc;                 // attenuation
        v = stat;
      }
      feat[idx] = (_Float16)v;
    }
#if HAVE_TDM
    __builtin_amdgcn_s_wait_tensorcnt(0);
#endif
    __syncthreads();
    v8f acc = {};
#pragma unroll 3
    for (int ks = 0; ks < 21; ++ks) {
      v16h a = load_a_frag(feat, 672, ks * 32, lane);
      v16h b = load_b_frag(bsh, 672, ks * 32, lane);
      acc = WMMA_F16(a, b, acc);
    }
    int g = lane & 15, hi = lane >> 4;
    if (g < F_OUT) {
      float bias = postb[t * F_OUT + g];
#pragma unroll
      for (int v = 0; v < 8; ++v) {
        int row = tile * 16 + v + 8 * hi;
        yf16[row * 64 + t * F_OUT + g] = (_Float16)(acc[v] + bias);
      }
    }
    __syncthreads();
  }
  for (int idx = lane; idx < 16 * 14; idx += 32) {     // zero pad cols 50..63
    int nl = idx / 14, c = D + idx % 14;
    yf16[(tile * 16 + nl) * 64 + c] = (_Float16)0.f;
  }
}

// ---------- batch norm ----------
__global__ void k_bn_reduce(const float* __restrict__ htmp, float* __restrict__ bns) {
  int col = threadIdx.x & 63;
  int sub = threadIdx.x >> 6;
  float s = 0.f, s2 = 0.f;
  for (int node = blockIdx.x * 4 + sub; node < N_NODES; node += gridDim.x * 4) {
    float v = htmp[node * 64 + col];
    s += v; s2 += v * v;
  }
  if (col < D) { atomicAdd(&bns[col], s); atomicAdd(&bns[64 + col], s2); }
}
__global__ void k_bn_apply(const float* __restrict__ htmp, const float* __restrict__ bns,
                           const float* __restrict__ gamma, const float* __restrict__ beta,
                           float* __restrict__ h, _Float16* __restrict__ hf16) {
  int i = blockIdx.x * 256 + threadIdx.x;
  if (i >= N_NODES * 64) return;
  int c = i & 63;
  float v = 0.f;
  if (c < D) {
    float mu = bns[c] * (1.f / N_NODES);
    float var = bns[64 + c] * (1.f / N_NODES) - mu * mu;
    v = gamma[c] * (htmp[i] - mu) * rsqrtf(var + 1e-5f) + beta[c];
    v = fmaxf(v, 0.f);
  }
  h[i] = v; hf16[i] = (_Float16)v;
}

// ---------- final MLP + sigmoid ----------
__global__ void k_mlp(const float* __restrict__ h, const float* __restrict__ W1,
                      const float* __restrict__ b1, const float* __restrict__ W2,
                      const float* __restrict__ b2, float* __restrict__ out) {
  int n = blockIdx.x * 256 + threadIdx.x;
  if (n >= N_NODES) return;
  float acc = b2[0];
  for (int j = 0; j < 25; ++j) {
    float s = b1[j];
    for (int i = 0; i < D; ++i) s += h[n * 64 + i] * W1[j * D + i];
    acc += fmaxf(s, 0.f) * W2[j];
  }
  out[n] = 1.f / (1.f + expf(-acc));
}

extern "C" void kernel_launch(void* const* d_in, const int* in_sizes, int n_in,
                              void* d_out, int out_size, void* d_ws, size_t ws_size,
                              hipStream_t stream) {
  const int*   x        = (const int*)d_in[0];
  const int*   ei       = (const int*)d_in[1];
  const float* emb      = (const float*)d_in[4];
  const float* pre_W    = (const float*)d_in[5];
  const float* pre_b    = (const float*)d_in[6];
  const float* post_W   = (const float*)d_in[7];
  const float* post_b   = (const float*)d_in[8];
  const float* lin_W    = (const float*)d_in[9];
  const float* lin_b    = (const float*)d_in[10];
  const float* bn_gamma = (const float*)d_in[11];
  const float* bn_beta  = (const float*)d_in[12];
  const float* mlp_W1   = (const float*)d_in[13];
  const float* mlp_b1   = (const float*)d_in[14];
  const float* mlp_W2   = (const float*)d_in[15];
  const float* mlp_b2   = (const float*)d_in[16];
  float* out = (float*)d_out;

  char* base = (char*)d_ws;
  size_t off = 0;
  auto carve = [&](size_t bytes) -> void* {
    void* p = base + off;
    off = (off + bytes + 255) & ~(size_t)255;
    return p;
  };
  float*    deg    = (float*)carve((size_t)N_NODES * 4);
  float*    avgl   = (float*)carve(256);
  int*      rowptr = (int*)carve((size_t)(N_NODES + 1) * 4);
  int*      cursor = (int*)carve((size_t)N_NODES * 4);
  int*      csrsrc = (int*)carve((size_t)N_EDGES * 4);
  float*    h      = (float*)carve((size_t)N_NODES * 64 * 4);
  _Float16* hf16   = (_Float16*)carve((size_t)N_NODES * 64 * 2);
  _Float16* yf16   = (_Float16*)carve((size_t)N_NODES * 64 * 2);
  float*    htmp   = (float*)carve((size_t)N_NODES * 64 * 4);
  float*    aiaj   = (float*)carve((size_t)N_NODES * 512 * 4);
  float*    ssum   = (float*)carve((size_t)N_NODES * 256 * 4);
  float*    ssq    = (float*)carve((size_t)N_NODES * 256 * 4);
  float*    smn    = (float*)carve((size_t)N_NODES * 256 * 4);
  float*    smx    = (float*)carve((size_t)N_NODES * 256 * 4);
  _Float16* Bpre   = (_Float16*)carve((size_t)N_LAYERS * 512 * 64 * 2);
  _Float16* Bpost  = (_Float16*)carve((size_t)N_LAYERS * TOWERS * 16 * 672 * 2);
  _Float16* Blin   = (_Float16*)carve((size_t)N_LAYERS * 64 * 64 * 2);
  float*    bns    = (float*)carve(128 * 4);

  auto cdiv = [](int a, int b) { return (a + b - 1) / b; };

  k_zero_f32<<<cdiv(N_NODES, 256), 256, 0, stream>>>(deg, N_NODES);
  k_zero_f32<<<1, 256, 0, stream>>>(avgl, 1);
  k_degree<<<cdiv(N_EDGES, 256), 256, 0, stream>>>(ei, deg);
  k_avglog<<<cdiv(N_NODES, 256), 256, 0, stream>>>(deg, avgl);
  k_scan_rowptr<<<1, 256, 0, stream>>>(deg, rowptr, cursor);
  k_build_csr<<<cdiv(N_EDGES, 256), 256, 0, stream>>>(ei, cursor, csrsrc);
  k_embed<<<cdiv(N_NODES * 64, 256), 256, 0, stream>>>(x, emb, h, hf16);
  k_prep_pre<<<cdiv(N_LAYERS * 512 * 64, 256), 256, 0, stream>>>(pre_W, Bpre);
  k_prep_post<<<cdiv(N_LAYERS * TOWERS * 16 * 672, 256), 256, 0, stream>>>(post_W, Bpost);
  k_prep_lin<<<cdiv(N_LAYERS * 64 * 64, 256), 256, 0, stream>>>(lin_W, Blin);

  for (int l = 0; l < N_LAYERS; ++l) {
    dim3 gpre(N_NODES / 16, 512 / 64);
    k_gemm_nk64<<<gpre, 32, 0, stream>>>(hf16, Bpre + (size_t)l * 512 * 64, aiaj, 512,
                                         (const float*)nullptr, 0);
    k_aggregate<<<N_NODES, 256, 0, stream>>>(rowptr, csrsrc, aiaj, pre_b + l * TOWERS * D,
                                             ssum, ssq, smn, smx);
    k_finalize<<<N_NODES / 16, 32, 0, stream>>>(h, ssum, ssq, smn, smx, deg, avgl,
                                                Bpost + (size_t)l * TOWERS * 16 * 672,
                                                post_b + l * TOWERS * F_OUT, yf16);
    dim3 glin(N_NODES / 16, 1);
    k_gemm_nk64<<<glin, 32, 0, stream>>>(yf16, Blin + (size_t)l * 64 * 64, htmp, 64,
                                         lin_b + l * D, D);
    k_zero_f32<<<1, 256, 0, stream>>>(bns, 128);
    k_bn_reduce<<<256, 256, 0, stream>>>(htmp, bns);
    k_bn_apply<<<cdiv(N_NODES * 64, 256), 256, 0, stream>>>(htmp, bns, bn_gamma + l * D,
                                                            bn_beta + l * D, h, hf16);
  }
  k_mlp<<<cdiv(N_NODES, 256), 256, 0, stream>>>(h, mlp_W1, mlp_b1, mlp_W2, mlp_b2, out);
}
